// Back_incor_18425409699927
// MI455X (gfx1250) — compile-verified, matrix operand order checked by reading, not run on hardware
//
#include <hip/hip_runtime.h>
#include <hip/hip_bf16.h>

// ---------------------------------------------------------------------------
// Sizes
// ---------------------------------------------------------------------------
#define BT    512          // 16*32
#define NROW  100          // local rows
#define NPAD  112          // padded to 7*16
#define MROW  49           // bins
#define MPAD  64           // padded to 4*16
#define CDIM  1024
#define AFFIN 1073         // C + BINS
#define AFFPAD 1088        // padded to mult of 64 halves

typedef __attribute__((ext_vector_type(16))) _Float16 v16h;
typedef __attribute__((ext_vector_type(8)))  float    v8f;
typedef __attribute__((ext_vector_type(4)))  unsigned int u32x4;
typedef __attribute__((ext_vector_type(4)))  int i32x4;
typedef __attribute__((ext_vector_type(8)))  int i32x8;

union FragU { uint4 u[2]; v16h h; };

// Fragment loader for X @ Y^T GEMMs: both operands stored row-major over K.
// ISA 16-bit A-matrix 16x32 layout: lane l -> row (l&15); g=l>>4;
// VGPR0-3 = halves [8g..8g+7], VGPR4-7 = halves [16+8g..16+8g+7].
__device__ __forceinline__ v16h load_frag(const _Float16* rowBaseAtK, int lane) {
    FragU f;
    const uint4* p = reinterpret_cast<const uint4*>(rowBaseAtK + ((lane >> 4) << 3));
    f.u[0] = p[0];   // halves [8g .. 8g+7]
    f.u[1] = p[2];   // halves [16+8g .. 16+8g+7]
    return f.h;
}

// ---------------------------------------------------------------------------
// Tensor Data Mover staging of a [rows x cols] f16 tile (row stride in elems)
// into LDS at byte offset ldsByteOff. Descriptor packing per ISA 08_async
// sections 8.3 (group0) / 8.4 (group1); 2-D tensor so groups 2/3 are zero.
// Issued by one wave; caller barriers before use.
// ---------------------------------------------------------------------------
#if __has_builtin(__builtin_amdgcn_tensor_load_to_lds) && \
    __has_builtin(__builtin_amdgcn_s_wait_tensorcnt)
#define HAVE_TDM 1
#else
#define HAVE_TDM 0
#endif

__device__ __forceinline__ void tdm_stage_tile(unsigned ldsByteOff, const _Float16* g,
                                               unsigned rows, unsigned cols,
                                               unsigned strideElems) {
#if HAVE_TDM
    unsigned long long ga = (unsigned long long)(size_t)g;
    u32x4 g0;
    g0[0] = 1u;                                         // count=1, user, no gather
    g0[1] = ldsByteOff;                                 // lds_addr (bytes)
    g0[2] = (unsigned)(ga & 0xffffffffu);               // global_addr[31:0]
    g0[3] = (unsigned)((ga >> 32) & 0x01ffffffu)        // global_addr[56:32]
            | (2u << 30);                               // type = 2 ("image")
    i32x8 g1;
    g1[0] = (int)(1u << 16);                            // data_size=1 (2 bytes)
    g1[1] = (int)((cols & 0xffffu) << 16);              // tensor_dim0[15:0]
    g1[2] = (int)((cols >> 16) | ((rows & 0xffffu) << 16)); // td0[31:16]|td1[15:0]
    g1[3] = (int)((rows >> 16) | ((cols & 0xffffu) << 16)); // td1[31:16]|tile_dim0
    g1[4] = (int)(rows & 0xffffu);                      // tile_dim1 (tile_dim2=0)
    g1[5] = (int)strideElems;                           // tensor_dim0_stride[31:0]
    g1[6] = 0;
    g1[7] = 0;
    i32x4 z4 = {0, 0, 0, 0};
#if defined(__clang_major__) && __clang_major__ >= 23
    i32x8 z8 = {0, 0, 0, 0, 0, 0, 0, 0};
    __builtin_amdgcn_tensor_load_to_lds(g0, g1, z4, z4, z8, 0);
#else
    __builtin_amdgcn_tensor_load_to_lds(g0, g1, z4, z4, 0);
#endif
    __builtin_amdgcn_s_wait_tensorcnt(0);
#else
    (void)ldsByteOff; (void)g; (void)rows; (void)cols; (void)strideElems;
#endif
}

// Stage a 16 x CDIM f16 tile into dynamic LDS (offset 0) and barrier.
__device__ __forceinline__ void stage_x_tile(_Float16* xlds, const _Float16* gX) {
#if HAVE_TDM
    if ((threadIdx.x >> 5) == 0) tdm_stage_tile(0u, gX, 16u, CDIM, CDIM);
#else
    for (unsigned i = threadIdx.x; i < 16u * CDIM / 8u; i += blockDim.x)
        ((uint4*)xlds)[i] = ((const uint4*)gX)[i];
#endif
    __syncthreads();
}

// ---------------------------------------------------------------------------
// K0: convert weights to f16 (W_adj [C][C]; W_aff [C][1073] -> padded [C][1088]
//     plus transposed tail Waff2t [C][MPAD])
// ---------------------------------------------------------------------------
__global__ void k_weights(const float* __restrict__ Wadj, const float* __restrict__ Waff,
                          _Float16* __restrict__ WadjH, _Float16* __restrict__ WaffH,
                          _Float16* __restrict__ Waff2t) {
    int d = blockIdx.x, tid = threadIdx.x;
    for (int c = tid; c < CDIM; c += 256)
        WadjH[(size_t)d * CDIM + c] = (_Float16)Wadj[(size_t)d * CDIM + c];
    for (int j = tid; j < AFFPAD; j += 256)
        WaffH[(size_t)d * AFFPAD + j] = (_Float16)(j < AFFIN ? Waff[(size_t)d * AFFIN + j] : 0.f);
    if (tid < MPAD)
        Waff2t[(size_t)d * MPAD + tid] =
            (_Float16)(tid < MROW ? Waff[(size_t)d * AFFIN + CDIM + tid] : 0.f);
}

// ---------------------------------------------------------------------------
// K1: L2-normalize rows + f16 convert, zero-padded rows.
//     grid = (BT, NPAD + MPAD); y<NPAD -> local rows; else global rows.
// ---------------------------------------------------------------------------
__global__ void k_norm(const float* __restrict__ lf, const float* __restrict__ gf,
                       _Float16* __restrict__ nl, _Float16* __restrict__ ng,
                       _Float16* __restrict__ gfh) {
    int b = blockIdx.x, y = blockIdx.y, tid = threadIdx.x;
    __shared__ float red[8];
    const float* src = nullptr;
    _Float16 *d0, *d1 = nullptr;
    bool valid;
    if (y < NPAD) {
        d0 = nl + ((size_t)b * NPAD + y) * CDIM;
        valid = (y < NROW);
        if (valid) src = lf + ((size_t)b * NROW + y) * CDIM;
    } else {
        int m = y - NPAD;
        d0 = ng  + ((size_t)b * MPAD + m) * CDIM;
        d1 = gfh + ((size_t)b * MPAD + m) * CDIM;
        valid = (m < MROW);
        if (valid) src = gf + ((size_t)b * MROW + m) * CDIM;
    }
    float vals[4];
    float ss = 0.f;
    if (valid) {
        for (int i = 0; i < 4; i++) { vals[i] = src[tid + 256 * i]; ss += vals[i] * vals[i]; }
    } else {
        for (int i = 0; i < 4; i++) vals[i] = 0.f;
    }
    for (int o = 16; o > 0; o >>= 1) ss += __shfl_xor(ss, o, 32);
    if ((tid & 31) == 0) red[tid >> 5] = ss;
    __syncthreads();
    float tot = red[0] + red[1] + red[2] + red[3] + red[4] + red[5] + red[6] + red[7];
    float scale = valid ? 1.f / fmaxf(sqrtf(tot), 1e-12f) : 0.f;
    for (int i = 0; i < 4; i++) d0[tid + 256 * i] = (_Float16)(vals[i] * scale);
    if (d1) for (int i = 0; i < 4; i++) d1[tid + 256 * i] = (_Float16)vals[i];
}

// ---------------------------------------------------------------------------
// K2/K3: out[m][d] (or transposed) = X[b][MPAD][C] @ Y[d][ystride]^T, K=C.
//     grid = (BT, MPAD/16, 16), block = 128 (4 waves, wave -> d-tile).
//     Shared 16xC X tile is TDM-staged into LDS once per block.
//     mode 0: store out[b][m][1024] (ngW);  mode 1: store out[b][d][64] (Gt)
// ---------------------------------------------------------------------------
__global__ void k_gemm64(const _Float16* __restrict__ X, const _Float16* __restrict__ Y,
                         int ystride, _Float16* __restrict__ out, int mode) {
    extern __shared__ _Float16 xlds[];   // 16 * CDIM halves = 32 KB
    int b = blockIdx.x, tm = blockIdx.y;
    int wave = threadIdx.x >> 5, lane = threadIdx.x & 31;
    int dt = blockIdx.z * 4 + wave;

    const _Float16* gX = X + (size_t)b * MPAD * CDIM + (size_t)(tm * 16) * CDIM;
    stage_x_tile(xlds, gX);

    const _Float16* xrow = xlds + (size_t)(lane & 15) * CDIM;
    const _Float16* yrow = Y + (size_t)(dt * 16 + (lane & 15)) * ystride;
    v8f acc = {};
    for (int kk = 0; kk < CDIM; kk += 32) {
        __builtin_prefetch(yrow + kk + 256, 0, 0);
        v16h a  = load_frag(xrow + kk, lane);      // ds_load_b128 x2
        v16h bf = load_frag(yrow + kk, lane);      // global_load_b128 x2
        acc = __builtin_amdgcn_wmma_f32_16x16x32_f16(false, a, false, bf, (short)0, acc,
                                                     false, false);
    }
    int g = lane >> 4, col = dt * 16 + (lane & 15);
    for (int v = 0; v < 8; v++) {
        int m = tm * 16 + v + 8 * g;
        _Float16 hv = (_Float16)acc[v];
        if (mode == 0) out[(size_t)b * MPAD * CDIM + (size_t)m * CDIM + col] = hv;
        else           out[(size_t)b * CDIM * MPAD + (size_t)col * MPAD + m] = hv;
    }
}

// ---------------------------------------------------------------------------
// K4: A_raw[b][NPAD][MPAD] = nl @ ngW^T, K=C, f32 out.
//     grid = (BT, NPAD/16), block = 128 (wave -> m-tile of 4).
//     16xC nl tile TDM-staged into LDS (shared by all 4 waves).
// ---------------------------------------------------------------------------
__global__ void k_adj(const _Float16* __restrict__ nl, const _Float16* __restrict__ ngW,
                      float* __restrict__ Araw) {
    extern __shared__ _Float16 xlds[];   // 32 KB
    int b = blockIdx.x, tn = blockIdx.y;
    int wave = threadIdx.x >> 5, lane = threadIdx.x & 31;

    const _Float16* gX = nl + ((size_t)b * NPAD + tn * 16) * CDIM;
    stage_x_tile(xlds, gX);

    const _Float16* xrow = xlds + (size_t)(lane & 15) * CDIM;
    const _Float16* yrow = ngW + ((size_t)b * MPAD + wave * 16 + (lane & 15)) * CDIM;
    v8f acc = {};
    for (int kk = 0; kk < CDIM; kk += 32) {
        __builtin_prefetch(yrow + kk + 256, 0, 0);
        v16h a  = load_frag(xrow + kk, lane);
        v16h bf = load_frag(yrow + kk, lane);
        acc = __builtin_amdgcn_wmma_f32_16x16x32_f16(false, a, false, bf, (short)0, acc,
                                                     false, false);
    }
    int g = lane >> 4, col = wave * 16 + (lane & 15);
    for (int v = 0; v < 8; v++) {
        int n = tn * 16 + v + 8 * g;
        Araw[((size_t)b * NPAD + n) * MPAD + col] = acc[v];
    }
}

// ---------------------------------------------------------------------------
// K5: masked softmax(5*A_raw) over m<49 -> As f16; raw A_raw -> Ar f16.
//     grid = (BT, NPAD/8), block = 256 (wave -> one row; lanes cover m, m+32)
// ---------------------------------------------------------------------------
__global__ void k_softmax(const float* __restrict__ Araw, _Float16* __restrict__ As,
                          _Float16* __restrict__ Ar) {
    int b = blockIdx.x;
    int wave = threadIdx.x >> 5, lane = threadIdx.x & 31;
    int n = blockIdx.y * 8 + wave;
    const float* ar = Araw + ((size_t)b * NPAD + n) * MPAD;
    bool f0 = lane < MROW, f1 = (lane + 32) < MROW;
    float x0 = f0 ? ar[lane] : 0.f;
    float x1 = f1 ? ar[lane + 32] : 0.f;
    float mx = fmaxf(f0 ? x0 : -1e30f, f1 ? x1 : -1e30f);
    for (int o = 16; o > 0; o >>= 1) mx = fmaxf(mx, __shfl_xor(mx, o, 32));
    float e0 = f0 ? __expf(5.f * (x0 - mx)) : 0.f;
    float e1 = f1 ? __expf(5.f * (x1 - mx)) : 0.f;
    float s = e0 + e1;
    for (int o = 16; o > 0; o >>= 1) s += __shfl_xor(s, o, 32);
    float inv = 1.f / s;
    _Float16* asr = As + ((size_t)b * NPAD + n) * MPAD;
    _Float16* arr = Ar + ((size_t)b * NPAD + n) * MPAD;
    asr[lane]      = (_Float16)(e0 * inv);
    asr[lane + 32] = (_Float16)(e1 * inv);
    arr[lane]      = (_Float16)x0;
    arr[lane + 32] = (_Float16)x1;
}

// ---------------------------------------------------------------------------
// K6: y = As @ Gt^T + Ar @ Waff2t^T + bias  (K=64 each), then LayerNorm +
//     LeakyReLU over C=1024 via a 64 KB LDS row tile.
//     grid = (BT, NPAD/16), block = 256 (8 waves; wave -> 8 d-tiles, then 2 rows)
// ---------------------------------------------------------------------------
__global__ void k_final(const _Float16* __restrict__ As, const _Float16* __restrict__ Ar,
                        const _Float16* __restrict__ Gt, const _Float16* __restrict__ W2t,
                        const float* __restrict__ bias, const float* __restrict__ gam,
                        const float* __restrict__ bet, float* __restrict__ out) {
    __shared__ float yt[16 * CDIM];   // 64 KB
    int b = blockIdx.x, tn = blockIdx.y;
    int wave = threadIdx.x >> 5, lane = threadIdx.x & 31;
    const _Float16* asrow = As + ((size_t)b * NPAD + tn * 16 + (lane & 15)) * MPAD;
    const _Float16* arrow = Ar + ((size_t)b * NPAD + tn * 16 + (lane & 15)) * MPAD;
    int g = lane >> 4;
    for (int t = 0; t < 8; t++) {
        int dt = wave * 8 + t;
        const _Float16* y1 = Gt  + ((size_t)b * CDIM + dt * 16 + (lane & 15)) * MPAD;
        const _Float16* y2 = W2t + (size_t)(dt * 16 + (lane & 15)) * MPAD;
        v8f acc = {};
        for (int kk = 0; kk < MPAD; kk += 32) {
            v16h a  = load_frag(asrow + kk, lane);
            v16h bf = load_frag(y1 + kk, lane);
            acc = __builtin_amdgcn_wmma_f32_16x16x32_f16(false, a, false, bf, (short)0, acc,
                                                         false, false);
        }
        for (int kk = 0; kk < MPAD; kk += 32) {
            v16h a  = load_frag(arrow + kk, lane);
            v16h bf = load_frag(y2 + kk, lane);
            acc = __builtin_amdgcn_wmma_f32_16x16x32_f16(false, a, false, bf, (short)0, acc,
                                                         false, false);
        }
        int col = dt * 16 + (lane & 15);
        float bv = bias[col];
        for (int v = 0; v < 8; v++) yt[(v + 8 * g) * CDIM + col] = acc[v] + bv;
    }
    __syncthreads();
    for (int rr = 0; rr < 2; rr++) {
        int r = wave * 2 + rr;
        int n = tn * 16 + r;
        float s = 0.f, sq = 0.f;
        for (int c = lane; c < CDIM; c += 32) {
            float v = yt[r * CDIM + c];
            s += v; sq += v * v;
        }
        for (int o = 16; o > 0; o >>= 1) { s += __shfl_xor(s, o, 32); sq += __shfl_xor(sq, o, 32); }
        float mu = s * (1.f / CDIM);
        float var = sq * (1.f / CDIM) - mu * mu;
        float rs = rsqrtf(var + 1e-5f);
        if (n < NROW) {
            float* op = out + ((size_t)b * NROW + n) * CDIM;
            for (int c = lane; c < CDIM; c += 32) {
                float v = (yt[r * CDIM + c] - mu) * rs * gam[c] + bet[c];
                op[c] = v >= 0.f ? v : 0.01f * v;
            }
        }
    }
}

// ---------------------------------------------------------------------------
// Host launcher
// ---------------------------------------------------------------------------
extern "C" void kernel_launch(void* const* d_in, const int* in_sizes, int n_in,
                              void* d_out, int out_size, void* d_ws, size_t ws_size,
                              hipStream_t stream) {
    const float* lf   = (const float*)d_in[0];
    const float* gf   = (const float*)d_in[1];
    // d_in[2] = pos (unused by the reference computation)
    const float* Wadj = (const float*)d_in[3];
    const float* Waff = (const float*)d_in[4];
    const float* baff = (const float*)d_in[5];
    const float* lng  = (const float*)d_in[6];
    const float* lnb  = (const float*)d_in[7];
    float* out = (float*)d_out;

    char* ws = (char*)d_ws;
    size_t o = 0;
    auto alloc = [&](size_t bytes) { char* p = ws + o; o += (bytes + 255) & ~(size_t)255; return p; };
    _Float16* nlH    = (_Float16*)alloc((size_t)BT * NPAD * CDIM * 2);
    _Float16* ngH    = (_Float16*)alloc((size_t)BT * MPAD * CDIM * 2);
    _Float16* gfH    = (_Float16*)alloc((size_t)BT * MPAD * CDIM * 2);
    _Float16* WadjH  = (_Float16*)alloc((size_t)CDIM * CDIM * 2);
    _Float16* WaffH  = (_Float16*)alloc((size_t)CDIM * AFFPAD * 2);
    _Float16* Waff2t = (_Float16*)alloc((size_t)CDIM * MPAD * 2);
    _Float16* ngW    = (_Float16*)alloc((size_t)BT * MPAD * CDIM * 2);
    _Float16* GtH    = (_Float16*)alloc((size_t)BT * CDIM * MPAD * 2);
    float*    Araw   = (float*)   alloc((size_t)BT * NPAD * MPAD * 4);
    _Float16* AsH    = (_Float16*)alloc((size_t)BT * NPAD * MPAD * 2);
    _Float16* ArH    = (_Float16*)alloc((size_t)BT * NPAD * MPAD * 2);

    const size_t xTileBytes = (size_t)16 * CDIM * sizeof(_Float16);  // 32 KB

    k_weights<<<dim3(CDIM), dim3(256), 0, stream>>>(Wadj, Waff, WadjH, WaffH, Waff2t);
    k_norm<<<dim3(BT, NPAD + MPAD), dim3(256), 0, stream>>>(lf, gf, nlH, ngH, gfH);
    // ngW[b] = ng @ W_adj^T      (store row-major over d==c)
    k_gemm64<<<dim3(BT, MPAD / 16, 16), dim3(128), xTileBytes, stream>>>(ngH, WadjH, CDIM,
                                                                         ngW, 0);
    // Gt[b]  = (gf @ W_aff1^T)^T (store [d][m] so final GEMM's B is K-contiguous)
    k_gemm64<<<dim3(BT, MPAD / 16, 16), dim3(128), xTileBytes, stream>>>(gfH, WaffH, AFFPAD,
                                                                         GtH, 1);
    // A_raw[b] = nl @ ngW^T
    k_adj<<<dim3(BT, NPAD / 16), dim3(128), xTileBytes, stream>>>(nlH, ngW, Araw);
    k_softmax<<<dim3(BT, NPAD / 8), dim3(256), 0, stream>>>(Araw, AsH, ArH);
    k_final<<<dim3(BT, NPAD / 16), dim3(256), 0, stream>>>(AsH, ArH, GtH, Waff2t,
                                                           baff, lng, lnb, out);
}
